// MonotonicAttention_36172214567350
// MI455X (gfx1250) — matrix operand order repaired and checked by reading, not compiled, not run on hardware
//
#include <hip/hip_runtime.h>
#include <math.h>

typedef __bf16 bf16;
typedef bf16  v8bf  __attribute__((ext_vector_type(8)));
typedef bf16  v16bf __attribute__((ext_vector_type(16)));
typedef float v8f   __attribute__((ext_vector_type(8)));
typedef unsigned int u32x4 __attribute__((ext_vector_type(4)));
typedef int i32x4 __attribute__((ext_vector_type(4)));
typedef int i32x8 __attribute__((ext_vector_type(8)));

#define B_ 32
#define S_ 2048
#define D_ 512
#define A_ 256

// LDS row pitch for the staged W tile: 512 bf16 = 1024B + 16B pad (TDM pad:
// interval 256 dwords, amount 4 dwords) -> 1040B = 260 dwords -> bank stride 4.
#define WROWP 1040
#define WLDS_BYTES (A_ * WROWP)   // 266240 <= 320KB WGP LDS

// ---------------- K1a: weight-normed output vector: weff = g * v_w / ||v_w||
__global__ void prep_weff(const float* __restrict__ vw, const float* __restrict__ vg,
                          float* __restrict__ weff) {
    __shared__ float sh[256];
    int t = threadIdx.x;
    float v = vw[t];
    sh[t] = v * v;
    __syncthreads();
    for (int off = 128; off > 0; off >>= 1) {
        if (t < off) sh[t] += sh[t + off];
        __syncthreads();
    }
    float nrm = sqrtf(sh[0]);
    weff[t] = vg[0] * v / nrm;
}

// ---------------- K1b: base[b,a] = dec[b,:]·V[a,:] + bias[a]
__global__ void prep_base(const float* __restrict__ dec, const float* __restrict__ V,
                          const float* __restrict__ bias, float* __restrict__ base) {
    int b = blockIdx.x, a = threadIdx.x;
    const float* vr = V + (size_t)a * D_;
    const float* dr = dec + (size_t)b * D_;
    float s = 0.f;
    for (int d = 0; d < D_; ++d) s += vr[d] * dr[d];
    base[b * A_ + a] = s + bias[a];
}

// ---------------- K1c: W fp32 -> bf16 (B-matrix operand, L2-resident 256 KB)
__global__ void conv_w(const float* __restrict__ W, bf16* __restrict__ Wb) {
    int i = blockIdx.x * blockDim.x + threadIdx.x;
    if (i < A_ * D_) Wb[i] = (bf16)W[i];
}

// ---------------- K2: energy[b,s] = weff · tanh(enc @ W^T + base)  via bf16 WMMA
// One workgroup = 16 waves = 256 S-rows; W (bf16, 256 KB) staged in LDS once
// via the Tensor Data Mover, then every wave's B-fragments come from LDS.
__global__ void __launch_bounds__(512) energy_gemm(
    const float* __restrict__ enc, const bf16* __restrict__ Wb,
    const float* __restrict__ base, const float* __restrict__ weff,
    float* __restrict__ energy) {
    __shared__ char wlds[WLDS_BYTES];

    // ---- TDM: async DMA of the whole W[256x512] bf16 tile into LDS (one op
    // per workgroup), with per-row 16B padding for conflict-free ds reads.
    if (threadIdx.x == 0) {
        unsigned long long ga = (unsigned long long)(const void*)Wb;
        u32x4 g0;
        g0[0] = 1u;                                       // count=1, user mode
        g0[1] = 0u;                                       // lds_addr = 0
        g0[2] = (unsigned)(ga & 0xFFFFFFFFull);           // global_addr[31:0]
        g0[3] = (unsigned)((ga >> 32) & 0x01FFFFFFull)    // global_addr[56:32]
                | (2u << 30);                             // type = 2 ("image")
        i32x8 g1;
        g1[0] = (1 << 16)        // data_size = 1 -> 2 bytes (bf16)
              | (1 << 20)        // pad_enable
              | (7 << 22)        // pad_interval: 256 dwords (one 1024B row)
              | (3 << 25);       // pad_amount: 4 dwords (16B)
        g1[1] = (D_ & 0xFFFF) << 16;                       // tensor_dim0 lo16
        g1[2] = ((D_ >> 16) & 0xFFFF) | ((A_ & 0xFFFF) << 16); // dim0 hi, dim1 lo
        g1[3] = ((A_ >> 16) & 0xFFFF) | ((D_ & 0xFFFF) << 16); // dim1 hi, tile_dim0=512
        g1[4] = A_;                                        // tile_dim1=256, tile_dim2=0
        g1[5] = D_;                                        // tensor_dim0_stride lo32
        g1[6] = 0;                                         // stride hi, dim1_stride lo
        g1[7] = 0;
        i32x4 z4 = {0, 0, 0, 0};
#if __clang_major__ >= 23
        i32x8 z8 = {0, 0, 0, 0, 0, 0, 0, 0};
        __builtin_amdgcn_tensor_load_to_lds(g0, g1, z4, z4, z8, 0);
#else
        __builtin_amdgcn_tensor_load_to_lds(g0, g1, z4, z4, 0);
#endif
        __builtin_amdgcn_s_wait_tensorcnt(0);
    }
    __syncthreads();

    // Escape wlds's address through a throwaway copy so alias analysis must
    // assume the TDM DMA (invisible to it) wrote the buffer. The actual
    // fragment loads below go through `wlds` itself so address-space
    // inference still lowers them to ds_load (not flat_load).
    {
        char* esc = (char*)wlds;
        asm volatile("" : "+v"(esc) : : "memory");
    }

    const int wave = threadIdx.x >> 5;
    const int lane = threadIdx.x & 31;
    const int half = lane >> 4;   // which 16-lane half of the wave32
    const int m    = lane & 15;   // row (A-frag) / column (B-frag, C-frag)
    const int b    = blockIdx.y;
    const int s0   = blockIdx.x * 256 + wave * 16;

    const float* rp = enc + ((size_t)b * S_ + (size_t)(s0 + m)) * D_;

    v8f acc[16];
#pragma unroll
    for (int n = 0; n < 16; ++n) {
        v8f z = {0.f, 0.f, 0.f, 0.f, 0.f, 0.f, 0.f, 0.f};
        acc[n] = z;
    }

    // Per-lane base offset into the staged W tile (bytes).
    const int lofs = m * WROWP + (half * 8) * 2;

    for (int k0 = 0; k0 < D_; k0 += 32) {
        // A fragment (16x32 bf16): lanes 0-15 hold K={k0..k0+7, k0+16..k0+23},
        // lanes 16-31 hold K={k0+8..k0+15, k0+24..k0+31}; row = m.
        const int ka = k0 + half * 8;
        float4 f0 = *(const float4*)(rp + ka);
        float4 f1 = *(const float4*)(rp + ka + 4);
        float4 f2 = *(const float4*)(rp + ka + 16);
        float4 f3 = *(const float4*)(rp + ka + 20);
        if (k0 + 32 < D_) __builtin_prefetch(rp + k0 + 32 + half * 8, 0, 1);

        v16bf a;
        a[0]  = (bf16)f0.x; a[1]  = (bf16)f0.y; a[2]  = (bf16)f0.z; a[3]  = (bf16)f0.w;
        a[4]  = (bf16)f1.x; a[5]  = (bf16)f1.y; a[6]  = (bf16)f1.z; a[7]  = (bf16)f1.w;
        a[8]  = (bf16)f2.x; a[9]  = (bf16)f2.y; a[10] = (bf16)f2.z; a[11] = (bf16)f2.w;
        a[12] = (bf16)f3.x; a[13] = (bf16)f3.y; a[14] = (bf16)f3.z; a[15] = (bf16)f3.w;

        // B fragment (32x16 bf16) from LDS: column a = n*16+m, lanes 0-15
        // K={k0..k0+7, k0+16..k0+23}, lanes 16-31 K={k0+8..k0+15, k0+24..k0+31}.
#pragma unroll
        for (int n = 0; n < 16; ++n) {
            const int qo = lofs + k0 * 2 + n * 16 * WROWP;
            v8bf lo = *(const v8bf*)(wlds + qo);
            v8bf hi = *(const v8bf*)(wlds + qo + 32);
            v16bf bf = __builtin_shufflevector(lo, hi, 0, 1, 2, 3, 4, 5, 6, 7,
                                               8, 9, 10, 11, 12, 13, 14, 15);
            acc[n] = __builtin_amdgcn_wmma_f32_16x16x32_bf16(
                false, a, false, bf, (short)0, acc[n], false, false);
        }
    }

    // Epilogue: C layout -> lane holds column a = n*16+m, VGPR r -> row s0+half*8+r.
    float e[8] = {0.f, 0.f, 0.f, 0.f, 0.f, 0.f, 0.f, 0.f};
#pragma unroll
    for (int n = 0; n < 16; ++n) {
        const int aIdx = n * 16 + m;
        const float bs = base[b * A_ + aIdx];
        const float wv = weff[aIdx];
#pragma unroll
        for (int rr = 0; rr < 8; ++rr)
            e[rr] += tanhf(acc[n][rr] + bs) * wv;
    }
#pragma unroll
    for (int rr = 0; rr < 8; ++rr) {
        float v = e[rr];
        v += __shfl_xor(v, 1);
        v += __shfl_xor(v, 2);
        v += __shfl_xor(v, 4);
        v += __shfl_xor(v, 8);
        if (m == 0)
            energy[(size_t)b * S_ + (size_t)(s0 + half * 8 + rr)] = v;
    }
}

// ---------------- K3: monotonic attention recurrence (two blocked scans per batch)
__device__ __forceinline__ float block_scan_excl(float v, float* sh, int t) {
    float x = v;
    sh[t] = x;
    __syncthreads();
    for (int off = 1; off < 256; off <<= 1) {
        float y = (t >= off) ? sh[t - off] : 0.0f;
        __syncthreads();
        x += y;
        sh[t] = x;
        __syncthreads();
    }
    return x - v;  // exclusive prefix
}

__global__ void __launch_bounds__(256) mono_scan(
    const float* __restrict__ energy, const float* __restrict__ noise,
    const float* __restrict__ prev, const float* __restrict__ vbias,
    const float* __restrict__ rsc, float* __restrict__ alpha_out) {
    __shared__ float sh[256];
    const int b = blockIdx.x, t = threadIdx.x;
    const size_t base = (size_t)b * S_ + (size_t)t * 8;
    const float c0 = vbias[0] + rsc[0];

    float p[8], lc[8];
    float run = 0.f;
#pragma unroll
    for (int j = 0; j < 8; ++j) {
        float x = energy[base + j] + noise[base + j] + c0;
        float pj = 1.f / (1.f + expf(-x));
        p[j] = pj;
        float om = 1.f - pj;
        om = fminf(fmaxf(om, 1e-10f), 1.f);
        run += logf(om);
        lc[j] = run;  // local inclusive cumsum of log(1-p)
    }
    float off1 = block_scan_excl(run, sh, t);

    float cp[8], q[8];
    float run2 = 0.f;
#pragma unroll
    for (int j = 0; j < 8; ++j) {
        cp[j] = expf(off1 + lc[j]);          // cumprod(1-p), inclusive
        run2 += prev[base + j] / cp[j];
        q[j] = run2;                          // local inclusive cumsum
    }
    __syncthreads();
    float off2 = block_scan_excl(run2, sh, t);

#pragma unroll
    for (int j = 0; j < 8; ++j)
        alpha_out[base + j] = p[j] * cp[j] * (off2 + q[j]);
}

// ---------------- K4: context partials over S-chunks: part[b,c,d] = sum_s enc*alpha
__global__ void __launch_bounds__(256) ctx_partial(
    const float* __restrict__ enc, const float* __restrict__ alpha,
    float* __restrict__ part) {
    const int bx = blockIdx.x;        // b*8 + chunk
    const int b = bx >> 3, c = bx & 7;
    const int d = blockIdx.y * 256 + threadIdx.x;
    const float* ep = enc + ((size_t)b * S_ + (size_t)c * 256) * D_ + d;
    const float* ap = alpha + (size_t)b * S_ + (size_t)c * 256;
    float acc = 0.f;
    for (int s = 0; s < 256; ++s)
        acc += ep[(size_t)s * D_] * ap[s];
    part[(size_t)bx * D_ + d] = acc;
}

// ---------------- K5: reduce partials into context
__global__ void ctx_reduce(const float* __restrict__ part, float* __restrict__ ctx) {
    int i = blockIdx.x * blockDim.x + threadIdx.x;  // b*D + d
    if (i >= B_ * D_) return;
    int b = i / D_, d = i % D_;
    float s = 0.f;
    for (int c = 0; c < 8; ++c) s += part[(size_t)(b * 8 + c) * D_ + d];
    ctx[i] = s;
}

extern "C" void kernel_launch(void* const* d_in, const int* in_sizes, int n_in,
                              void* d_out, int out_size, void* d_ws, size_t ws_size,
                              hipStream_t stream) {
    (void)in_sizes; (void)n_in; (void)out_size; (void)ws_size;
    const float* enc   = (const float*)d_in[0];
    const float* dec   = (const float*)d_in[1];
    const float* prev  = (const float*)d_in[2];
    const float* noise = (const float*)d_in[3];
    const float* W     = (const float*)d_in[4];
    const float* V     = (const float*)d_in[5];
    const float* bias  = (const float*)d_in[6];
    const float* vw    = (const float*)d_in[7];
    const float* vg    = (const float*)d_in[8];
    const float* vbias = (const float*)d_in[9];
    const float* rsc   = (const float*)d_in[10];

    float* out   = (float*)d_out;
    float* ctx   = out;              // [B,D]   = 16384 floats
    float* alpha = out + B_ * D_;    // [B,S]   = 65536 floats

    char* ws = (char*)d_ws;
    float* weff   = (float*)(ws);                                   // 256 f
    float* base   = (float*)(ws + 1024);                            // 8192 f
    float* energy = (float*)(ws + 1024 + 32768);                    // 65536 f
    float* part   = (float*)(ws + 1024 + 32768 + 262144);           // 131072 f
    bf16*  Wb     = (bf16*) (ws + 1024 + 32768 + 262144 + 524288);  // 131072 bf16

    prep_weff<<<1, 256, 0, stream>>>(vw, vg, weff);
    prep_base<<<B_, 256, 0, stream>>>(dec, V, bias, base);
    conv_w<<<(A_ * D_ + 255) / 256, 256, 0, stream>>>(W, Wb);
    energy_gemm<<<dim3(S_ / 256, B_), 512, 0, stream>>>(enc, Wb, base, weff, energy);
    mono_scan<<<B_, 256, 0, stream>>>(energy, noise, prev, vbias, rsc, alpha);
    ctx_partial<<<dim3(B_ * 8, D_ / 256), 256, 0, stream>>>(enc, alpha, part);
    ctx_reduce<<<(B_ * D_ + 255) / 256, 256, 0, stream>>>(part, ctx);
}
